// GNNActorVariablePrice_24326694764553
// MI455X (gfx1250) — matrix operand order repaired and checked by reading, not compiled, not run on hardware
//
#include <hip/hip_runtime.h>
#include <hip/hip_bf16.h>

// ---------------------------------------------------------------------------
// GNNActorVariablePrice for MI455X (gfx1250): bf16 WMMA pipeline.
// Dominant cost: bilinear head = 10000 x 16384 x 256 GEMM (~84 GFLOP),
// fused so the [N, R*M] intermediate is never materialized.
// Round 2: bilinear loop order r -> jt -> rt -> k with B fragments held in
// registers across the 4 row-tiles (4x less L2 weight traffic), capped
// unrolling to stay below the 256-VGPR MSB boundary.
// ---------------------------------------------------------------------------

#define NN 10000      // nodes
#define EE 160000     // edges
#define CD 256        // channels (C == M == 256)
#define RR 64         // bilinear regions
#define NPAD 10048    // nodes padded to multiple of 64 for the bilinear tile

typedef __bf16 bf16;
typedef __attribute__((ext_vector_type(16))) __bf16 v16bf;
typedef __attribute__((ext_vector_type(8)))  __bf16 v8bf;
typedef __attribute__((ext_vector_type(8)))  float  v8f;

union BFrag { v16bf v; v8bf h[2]; };

__device__ __forceinline__ v8f wmma_bf16(const BFrag& a, const BFrag& b, v8f c) {
  // D(16x16 f32) = A(16x32 bf16) x B(32x16 bf16) + C
  return __builtin_amdgcn_wmma_f32_16x16x32_bf16(false, a.v, false, b.v,
                                                 (short)0, c, false, false);
}

// ---------------------------------------------------------------------------
// Elementwise helpers
// ---------------------------------------------------------------------------
__global__ void k_f32_to_bf16(const float* __restrict__ in, bf16* __restrict__ out, int n) {
  int i = blockIdx.x * blockDim.x + threadIdx.x;
  if (i < n) out[i] = (bf16)in[i];
}

__global__ void k_init(float* __restrict__ agg, float* __restrict__ deg,
                       bf16* __restrict__ x2pad, int nagg, int ndeg, int npad) {
  int i = blockIdx.x * blockDim.x + threadIdx.x;
  if (i < nagg) agg[i] = 0.0f;
  if (i < ndeg) deg[i] = 1.0f;           // +1 self loop
  if (i < npad) x2pad[i] = (bf16)0.0f;   // zero bilinear tail rows
}

__global__ void k_degree(const int* __restrict__ dst, float* __restrict__ deg, int e) {
  int i = blockIdx.x * blockDim.x + threadIdx.x;
  if (i < e) atomicAdd(&deg[dst[i]], 1.0f);
}

__global__ void k_rsqrt(const float* __restrict__ deg, float* __restrict__ dinv, int n) {
  int i = blockIdx.x * blockDim.x + threadIdx.x;
  if (i < n) dinv[i] = rsqrtf(deg[i]);
}

// one block (256 threads) per edge: agg[dst,c] += dinv[s]*dinv[d]*xw[src,c]
__global__ __launch_bounds__(256) void k_scatter(const int* __restrict__ src,
                                                 const int* __restrict__ dst,
                                                 const float* __restrict__ dinv,
                                                 const float* __restrict__ xw,
                                                 float* __restrict__ agg) {
  int e = blockIdx.x;
  int s = src[e], d = dst[e];
  float nrm = dinv[s] * dinv[d];
  int c = threadIdx.x;
  atomicAdd(&agg[d * CD + c], nrm * xw[s * CD + c]);
}

// x0 = relu(agg + dinv^2 * xw + conv_b) + x  -> bf16
__global__ void k_postconv(const float* __restrict__ agg, const float* __restrict__ xw,
                           const float* __restrict__ dinv, const float* __restrict__ conv_b,
                           const float* __restrict__ x, bf16* __restrict__ x0b, int total) {
  int i = blockIdx.x * blockDim.x + threadIdx.x;
  if (i >= total) return;
  int n = i >> 8, c = i & 255;
  float di = dinv[n];
  float v = agg[i] + di * di * xw[i] + conv_b[c];
  v = fmaxf(v, 0.0f) + x[i];
  x0b[i] = (bf16)v;
}

// ---------------------------------------------------------------------------
// Generic 256-wide GEMM: out[N,256] = act(A[N,256] @ W[256,256]^T + bias)
// A, W bf16 row-major. Block = 8 waves; wave w covers columns [32w, 32w+32),
// block b covers rows [16b, 16b+16). K loop fully unrolled: 16 WMMAs/wave.
// ---------------------------------------------------------------------------
template <bool BIAS, bool RELU, bool BF16OUT>
__global__ __launch_bounds__(256) void k_gemm256(const bf16* __restrict__ A,
                                                 const bf16* __restrict__ W,
                                                 const float* __restrict__ bias,
                                                 void* __restrict__ outp) {
  const int lane = threadIdx.x & 31;
  const int wave = threadIdx.x >> 5;
  const int half = lane >> 4, l15 = lane & 15;
  const int row0 = blockIdx.x * 16;
  const int colBase = wave * 32;

  const bf16* Arow = A + (size_t)(row0 + l15) * CD;
  const bf16* Wr0  = W + (size_t)(colBase + l15) * CD + half * 16;
  const bf16* Wr1  = Wr0 + 16 * CD;

  v8f acc0 = {}; v8f acc1 = {};
#pragma unroll
  for (int k = 0; k < CD; k += 32) {
    BFrag a, b0, b1;
    a.h[0]  = *(const v8bf*)(Arow + k + half * 16);
    a.h[1]  = *(const v8bf*)(Arow + k + half * 16 + 8);
    b0.h[0] = *(const v8bf*)(Wr0 + k);
    b0.h[1] = *(const v8bf*)(Wr0 + k + 8);
    b1.h[0] = *(const v8bf*)(Wr1 + k);
    b1.h[1] = *(const v8bf*)(Wr1 + k + 8);
    acc0 = wmma_bf16(a, b0, acc0);
    acc1 = wmma_bf16(a, b1, acc1);
  }

  float bia0 = BIAS ? bias[colBase + l15] : 0.0f;
  float bia1 = BIAS ? bias[colBase + 16 + l15] : 0.0f;
#pragma unroll
  for (int v = 0; v < 8; ++v) {
    int row = row0 + half * 8 + v;
    float f0 = acc0[v] + bia0;
    float f1 = acc1[v] + bia1;
    if (RELU) { f0 = fmaxf(f0, 0.0f); f1 = fmaxf(f1, 0.0f); }
    if (BF16OUT) {
      bf16* o = (bf16*)outp;
      o[(size_t)row * CD + colBase + l15]      = (bf16)f0;
      o[(size_t)row * CD + colBase + 16 + l15] = (bf16)f1;
    } else {
      float* o = (float*)outp;
      o[(size_t)row * CD + colBase + l15]      = f0;
      o[(size_t)row * CD + colBase + 16 + l15] = f1;
    }
  }
}

// x1[n] = a1[n,:] . w3 + b3 ; one wave per node, 8 nodes per block
__global__ __launch_bounds__(256) void k_lin3(const float* __restrict__ a1,
                                              const float* __restrict__ w3,
                                              const float* __restrict__ b3,
                                              float* __restrict__ out) {
  int lane = threadIdx.x & 31;
  int wave = threadIdx.x >> 5;
  int node = blockIdx.x * 8 + wave;
  if (node >= NN) return;
  const float* rowp = a1 + (size_t)node * CD;
  float s = 0.0f;
#pragma unroll
  for (int i = 0; i < 8; ++i) s += rowp[lane + 32 * i] * w3[lane + 32 * i];
#pragma unroll
  for (int off = 16; off >= 1; off >>= 1) s += __shfl_xor(s, off, 32);
  if (lane == 0) out[node] = s + b3[0];
}

// ---------------------------------------------------------------------------
// Fused bilinear head: out2[n,r] = x2[n] @ W[r] @ x2[n] + b[r]
// Block: 64-node tile of x2 in LDS (bf16, padded stride 264 -> 4-bank rotate).
// Wave w handles regions r = 8w..8w+7.
// Loop order r -> jt -> rt -> k: the full-K B fragment set for one (r, jt)
// column tile (64 VGPRs) is loaded once and reused across 4 row-tiles,
// cutting L2 weight traffic 4x (~1.3 GB total; weights L2-resident @8.4 MB).
// ---------------------------------------------------------------------------
#define LDSLD 264

__global__ __launch_bounds__(256) void k_bilinear(const bf16* __restrict__ x2b,
                                                  const bf16* __restrict__ Wb,
                                                  const float* __restrict__ bb,
                                                  float* __restrict__ out2) {
  __shared__ __align__(16) bf16 xs[64 * LDSLD];
  const int rowBase = blockIdx.x * 64;

  // cooperative tile load: 64 rows x 256 cols bf16, 16B chunks
  for (int i = threadIdx.x; i < 64 * 32; i += 256) {
    int r = i >> 5, cc = (i & 31) << 3;
    *(v8bf*)(&xs[r * LDSLD + cc]) = *(const v8bf*)(x2b + (size_t)(rowBase + r) * CD + cc);
  }
  __syncthreads();

  const int lane = threadIdx.x & 31;
  const int wave = threadIdx.x >> 5;
  const int half = lane >> 4, l15 = lane & 15;

#pragma unroll 1
  for (int ri = 0; ri < 8; ++ri) {
    const int r = wave * 8 + ri;
    const bf16* Wr = Wb + (size_t)r * (CD * CD);
    float acc[4][8] = {};

#pragma unroll 2
    for (int jt = 0; jt < 16; ++jt) {
      // B fragments for the whole K=256 of this (r, jt) column tile
      BFrag Breg[8];
      const bf16* Wn = Wr + (size_t)(jt * 16 + l15) * CD + half * 16;
#pragma unroll
      for (int k8 = 0; k8 < 8; ++k8) {
        Breg[k8].h[0] = *(const v8bf*)(Wn + k8 * 32);
        Breg[k8].h[1] = *(const v8bf*)(Wn + k8 * 32 + 8);
      }
      const int col = jt * 16 + l15;

#pragma unroll
      for (int rt = 0; rt < 4; ++rt) {
        const bf16* As = &xs[(rt * 16 + l15) * LDSLD];
        v8f d0 = {}; v8f d1 = {};
#pragma unroll
        for (int k8 = 0; k8 < 8; k8 += 2) {
          BFrag a;
          a.h[0] = *(const v8bf*)(As + k8 * 32 + half * 16);
          a.h[1] = *(const v8bf*)(As + k8 * 32 + half * 16 + 8);
          d0 = wmma_bf16(a, Breg[k8], d0);
          a.h[0] = *(const v8bf*)(As + k8 * 32 + 32 + half * 16);
          a.h[1] = *(const v8bf*)(As + k8 * 32 + 32 + half * 16 + 8);
          d1 = wmma_bf16(a, Breg[k8 + 1], d1);
        }
#pragma unroll
        for (int v = 0; v < 8; ++v) {
          float m = (float)xs[(rt * 16 + half * 8 + v) * LDSLD + col];
          acc[rt][v] += (d0[v] + d1[v]) * m;
        }
      }
    }

    // reduce the 16 column-lanes of each half and store this region's column
    float bias = bb[r];
#pragma unroll
    for (int rt = 0; rt < 4; ++rt) {
#pragma unroll
      for (int v = 0; v < 8; ++v) {
        float s = acc[rt][v];
        s += __shfl_xor(s, 8, 16);
        s += __shfl_xor(s, 4, 16);
        s += __shfl_xor(s, 2, 16);
        s += __shfl_xor(s, 1, 16);
        acc[rt][v] = s;
      }
      if (l15 == 0) {
#pragma unroll
        for (int v = 0; v < 8; ++v) {
          int node = rowBase + rt * 16 + half * 8 + v;
          if (node < NN) out2[(size_t)node * RR + r] = acc[rt][v] + bias;
        }
      }
    }
  }
}

// ---------------------------------------------------------------------------
// Launcher
// ---------------------------------------------------------------------------
extern "C" void kernel_launch(void* const* d_in, const int* in_sizes, int n_in,
                              void* d_out, int out_size, void* d_ws, size_t ws_size,
                              hipStream_t stream) {
  const float* x       = (const float*)d_in[0];
  const int*   ei      = (const int*)  d_in[1];
  const float* conv_w  = (const float*)d_in[2];
  const float* conv_b  = (const float*)d_in[3];
  const float* lin1_w  = (const float*)d_in[4];
  const float* lin1_b  = (const float*)d_in[5];
  const float* lin2_w  = (const float*)d_in[6];
  const float* lin2_b  = (const float*)d_in[7];
  const float* lin3_w  = (const float*)d_in[8];
  const float* lin3_b  = (const float*)d_in[9];
  const float* lin4_w  = (const float*)d_in[10];
  const float* lin4_b  = (const float*)d_in[11];
  const float* bilin_w = (const float*)d_in[12];
  const float* bilin_b = (const float*)d_in[13];

  float* x1_out = (float*)d_out;            // [N]
  float* x2_out = (float*)d_out + NN;       // [N, R]

  // workspace carving
  char* ws = (char*)d_ws;
  size_t off = 0;
  auto carve = [&](size_t bytes) -> void* {
    void* p = ws + off;
    off = (off + bytes + 255) & ~(size_t)255;
    return p;
  };
  bf16*  xb       = (bf16*) carve((size_t)NN * CD * 2);
  bf16*  conv_wb  = (bf16*) carve((size_t)CD * CD * 2);
  bf16*  lin1_wb  = (bf16*) carve((size_t)CD * CD * 2);
  bf16*  lin2_wb  = (bf16*) carve((size_t)CD * CD * 2);
  bf16*  lin4_wb  = (bf16*) carve((size_t)CD * CD * 2);
  bf16*  bilin_wb = (bf16*) carve((size_t)RR * CD * CD * 2);
  float* xw       = (float*)carve((size_t)NN * CD * 4);
  float* agg      = (float*)carve((size_t)NN * CD * 4);
  float* deg      = (float*)carve((size_t)NN * 4);
  float* dinv     = (float*)carve((size_t)NN * 4);
  bf16*  x0b      = (bf16*) carve((size_t)NN * CD * 2);
  bf16*  h0b      = (bf16*) carve((size_t)NN * CD * 2);
  float* a1       = (float*)carve((size_t)NN * CD * 4);
  bf16*  x2b      = (bf16*) carve((size_t)NPAD * CD * 2);

  const int T = 256;
  auto blocks = [](int n, int t) { return (n + t - 1) / t; };

  // --- prep: bf16 conversions + zero/one init ---
  k_f32_to_bf16<<<blocks(NN * CD, T), T, 0, stream>>>(x, xb, NN * CD);
  k_f32_to_bf16<<<blocks(CD * CD, T), T, 0, stream>>>(conv_w, conv_wb, CD * CD);
  k_f32_to_bf16<<<blocks(CD * CD, T), T, 0, stream>>>(lin1_w, lin1_wb, CD * CD);
  k_f32_to_bf16<<<blocks(CD * CD, T), T, 0, stream>>>(lin2_w, lin2_wb, CD * CD);
  k_f32_to_bf16<<<blocks(CD * CD, T), T, 0, stream>>>(lin4_w, lin4_wb, CD * CD);
  k_f32_to_bf16<<<blocks(RR * CD * CD, T), T, 0, stream>>>(bilin_w, bilin_wb, RR * CD * CD);
  k_init<<<blocks(NN * CD, T), T, 0, stream>>>(agg, deg, x2b + (size_t)NN * CD,
                                               NN * CD, NN, (NPAD - NN) * CD);

  // --- GCN conv: xw = x @ conv_w^T (f32 out, bias/relu deferred) ---
  k_gemm256<false, false, false><<<NN / 16, 256, 0, stream>>>(xb, conv_wb, nullptr, xw);

  // --- degree / norm / scatter ---
  const int* src = ei;
  const int* dst = ei + EE;
  k_degree<<<blocks(EE, T), T, 0, stream>>>(dst, deg, EE);
  k_rsqrt<<<blocks(NN, T), T, 0, stream>>>(deg, dinv, NN);
  k_scatter<<<EE, 256, 0, stream>>>(src, dst, dinv, xw, agg);

  // --- x0 = relu(agg + dinv^2*xw + b) + x ---
  k_postconv<<<blocks(NN * CD, T), T, 0, stream>>>(agg, xw, dinv, conv_b, x, x0b, NN * CD);

  // --- lin1 / lin2 / lin3 / lin4 ---
  k_gemm256<true, true, true ><<<NN / 16, 256, 0, stream>>>(x0b, lin1_wb, lin1_b, h0b);
  k_gemm256<true, true, false><<<NN / 16, 256, 0, stream>>>(h0b, lin2_wb, lin2_b, a1);
  k_lin3<<<NN / 8, 256, 0, stream>>>(a1, lin3_w, lin3_b, x1_out);
  k_gemm256<true, true, true ><<<NN / 16, 256, 0, stream>>>(h0b, lin4_wb, lin4_b, x2b);

  // --- fused bilinear head ---
  k_bilinear<<<NPAD / 64, 256, 0, stream>>>(x2b, bilin_wb, bilin_b, x2_out);
}
